// quantizer_58360015618417
// MI455X (gfx1250) — matrix operand order repaired and checked by reading, not compiled, not run on hardware
//
#include <hip/hip_runtime.h>

// ---------------------------------------------------------------------------
// VQ quantizer for MI455X (gfx1250, wave32, WMMA, async global->LDS).
// Cross-term GEMM in bf16 WMMA (f32 accumulate); norms/argmin/gather/loss f32.
// Codebook B-fragments staged into LDS once per block via
// global_load_async_to_lds_b128 (double-buffered, ASYNCcnt-pipelined).
// ---------------------------------------------------------------------------

typedef __attribute__((ext_vector_type(16))) __bf16 v16bf;
typedef __attribute__((ext_vector_type(8)))  float  v8f;

#define Bn 32
#define Dn 256
#define Hn 32
#define Wn 32
#define Kn 1024
#define HWn (Hn * Wn)          // 1024
#define Nn  (Bn * HWn)         // 32768 tokens
#define OUT_ELEMS (Nn * Dn)    // 8388608
#define TOK_PER_BLOCK 128      // 8 waves x 16 tokens
#define KT  (Kn / 16)          // 64 code tiles
#define CCH (Dn / 32)          // 8 K-dim chunks of 32

#define CHUNK_KT 4                                  // code tiles per staged chunk
#define NCHUNK   (KT / CHUNK_KT)                    // 16
#define CHUNK_USHORT (CHUNK_KT * CCH * 32 * 16)     // 16384 bf16
#define CHUNK_BYTES  (CHUNK_USHORT * 2)             // 32768 B
#define COPIES_PER_THREAD (CHUNK_BYTES / (256 * 16))// 8 x b128 per thread

__device__ __forceinline__ unsigned short f2bf(float f) {
  union { float f; unsigned u; } v; v.f = f;
  unsigned r = v.u + 0x7FFFu + ((v.u >> 16) & 1u);  // round-to-nearest-even
  return (unsigned short)(r >> 16);
}

// CDNA5 async global->LDS copy (ASYNCcnt-tracked), inline asm per ISA 15.18.3.
__device__ __forceinline__ void async_b128(unsigned lds_off, const void* gaddr) {
  asm volatile("global_load_async_to_lds_b128 %0, %1, off"
               :: "v"(lds_off), "v"(gaddr) : "memory");
}
__device__ __forceinline__ void wait_async_le8() {
  asm volatile("s_wait_asynccnt 0x8" ::: "memory");   // chunk g landed, g+1 in flight
}
__device__ __forceinline__ void wait_async_0() {
  asm volatile("s_wait_asynccnt 0x0" ::: "memory");
}

__device__ __forceinline__ void issue_chunk(const char* cbB, int g,
                                            unsigned short* buf, int tid) {
  const char* src = cbB + (size_t)g * CHUNK_BYTES + tid * 16;
  unsigned dst = (unsigned)(uintptr_t)buf + (unsigned)(tid * 16);
#pragma unroll
  for (int r = 0; r < COPIES_PER_THREAD; ++r)
    async_b128(dst + r * 4096, src + r * 4096);
}

// ---------------------------------------------------------------------------
// Prep: codebook -> bf16 B-fragment layout + c_sq.
// B fragment (32x16 bf16): lane n<16 holds column n, j=0..15 -> K=j;
// lanes 16..31 hold column n-16 with K=16+j. 32 lanes x 16 bf16 per (kt,c).
// ---------------------------------------------------------------------------
__global__ void __launch_bounds__(256) vq_prep(const float* __restrict__ cb,
                                               unsigned short* __restrict__ cbB,
                                               float* __restrict__ csq) {
  const int kt  = blockIdx.x;
  const int tid = threadIdx.x;

  const int c = tid >> 5, l = tid & 31;
  const int k = kt * 16 + (l & 15);
  const int dbase = c * 32 + ((l < 16) ? 0 : 16);
  const float* src = cb + (size_t)k * Dn + dbase;
  unsigned short* dst = cbB + ((size_t)(kt * CCH + c) * 32 + l) * 16;
#pragma unroll
  for (int j = 0; j < 16; ++j) dst[j] = f2bf(src[j]);

  __shared__ float red[256];
  const int kl = tid >> 4, part = tid & 15;
  const float* row = cb + (size_t)(kt * 16 + kl) * Dn + part * 16;
  float s = 0.f;
#pragma unroll
  for (int j = 0; j < 16; ++j) s += row[j] * row[j];
  red[kl * 16 + part] = s;
  __syncthreads();
  if (tid < 16) {
    float t = 0.f;
#pragma unroll
    for (int p = 0; p < 16; ++p) t += red[tid * 16 + p];
    csq[kt * 16 + tid] = t;
  }
}

// ---------------------------------------------------------------------------
// Main: per block = 128 tokens (8 waves x 16). A fragments in regs (via LDS),
// B chunks DMA'd to LDS (double buffer), v_wmma_f32_16x16x32_bf16 with dual
// accumulators, shfl argmin, gather + loss.
// ---------------------------------------------------------------------------
__global__ void __launch_bounds__(256) vq_main(const float* __restrict__ x,
                                               const float* __restrict__ cb,
                                               const unsigned short* __restrict__ cbB,
                                               const float* __restrict__ csq,
                                               float* __restrict__ out,
                                               float* __restrict__ idxOut,
                                               float* __restrict__ lossPart) {
  __shared__ unsigned short ldsA[TOK_PER_BLOCK * Dn];     // 64 KB A fragments
  __shared__ unsigned short ldsB[2][CHUNK_USHORT];        // 2 x 32 KB B staging
  __shared__ float ldsCSQ[Kn];                            // 4 KB
  __shared__ float ldsPS[256];
  __shared__ float ldsXSQ[TOK_PER_BLOCK];
  __shared__ int   ldsIdx[TOK_PER_BLOCK];

  const int tid = threadIdx.x;
  const int t0  = blockIdx.x * TOK_PER_BLOCK;
  const int bI  = t0 / HWn;            // 128 | 1024 => whole block shares b
  const int hw0 = t0 % HWn;
  const size_t xbase = (size_t)bI * Dn * HWn + hw0;

  // Kick off DMA of first two B chunks; overlaps with x staging below.
  issue_chunk((const char*)cbB, 0, ldsB[0], tid);
  issue_chunk((const char*)cbB, 1, ldsB[1], tid);

  // ---- Phase 1: stage x as bf16 A-fragments in LDS + per-token |x|^2 + csq
  const int i  = tid & 127;            // token-in-block (fixed per thread)
  const int dg = tid >> 7;             // 0/1: even/odd d rows
  float sq = 0.f;
  for (int it = 0; it < Dn / 2; ++it) {
    const int d = it * 2 + dg;
    const float v = x[xbase + (size_t)d * HWn + i];
    sq += v * v;
    // A-fragment (16x32 bf16): lanes 0-15: K 0-15; 16-31: K 16-31
    const int w = i >> 4, m = i & 15;
    const int c = d >> 5, kk = d & 31;
    int l, j;
    if (kk < 16)      { l = m;      j = kk;      }
    else              { l = m + 16; j = kk - 16; }
    ldsA[((w * CCH + c) * 32 + l) * 16 + j] = f2bf(v);
  }
  for (int q = tid; q < Kn; q += 256) ldsCSQ[q] = csq[q];
  ldsPS[tid] = sq;
  __syncthreads();
  if (tid < TOK_PER_BLOCK) ldsXSQ[tid] = ldsPS[tid] + ldsPS[tid + 128];
  __syncthreads();

  // ---- Phase 2: WMMA K loop over staged chunks + running argmin
  const int w      = tid >> 5;         // wave id: owns tokens [16w, 16w+16)
  const int lane   = tid & 31;
  const int col    = lane & 15;        // code column within 16-tile
  const int rowoff = (lane < 16) ? 0 : 8;

  // Hoist this wave's 8 A fragments into registers (64 VGPRs).
  const v16bf* afr = (const v16bf*)(&ldsA[(size_t)w * CCH * 32 * 16]);
  v16bf Afr[CCH];
#pragma unroll
  for (int c = 0; c < CCH; ++c) Afr[c] = afr[c * 32 + lane];

  float xs[8];
#pragma unroll
  for (int v = 0; v < 8; ++v) xs[v] = ldsXSQ[w * 16 + rowoff + v];

  float minv[8];
  int   mini[8];
#pragma unroll
  for (int v = 0; v < 8; ++v) { minv[v] = 3.0e38f; mini[v] = 0; }

  for (int g = 0; g < NCHUNK; ++g) {
    if (g < NCHUNK - 1) wait_async_le8(); else wait_async_0();
    __syncthreads();                   // chunk g visible to all waves
    const v16bf* bfr = (const v16bf*)(&ldsB[g & 1][0]);

    for (int ktl = 0; ktl < CHUNK_KT; ++ktl) {
      const int kt = g * CHUNK_KT + ktl;
      v8f acc0 = {}, acc1 = {};        // dual accumulators: half the RAW chain
#pragma unroll
      for (int c = 0; c < CCH / 2; ++c) {
        v16bf b0 = bfr[(ktl * CCH + c) * 32 + lane];            // ds_load_b128 x2
        v16bf b1 = bfr[(ktl * CCH + c + CCH / 2) * 32 + lane];
        acc0 = __builtin_amdgcn_wmma_f32_16x16x32_bf16(
            false, Afr[c], false, b0, (short)0, acc0, false, false);
        acc1 = __builtin_amdgcn_wmma_f32_16x16x32_bf16(
            false, Afr[c + CCH / 2], false, b1, (short)0, acc1, false, false);
      }
      const float cs = ldsCSQ[kt * 16 + col];
      const int kcode = kt * 16 + col;
#pragma unroll
      for (int v = 0; v < 8; ++v) {
        const float dist = xs[v] - 2.0f * (acc0[v] + acc1[v]) + cs;
        if (dist < minv[v]) { minv[v] = dist; mini[v] = kcode; } // lowest k on tie
      }
    }
    __syncthreads();                   // all waves done reading buf[g&1]
    if (g + 2 < NCHUNK)
      issue_chunk((const char*)cbB, g + 2, ldsB[g & 1], tid);
  }

  // cross-lane argmin within each 16-lane half (rows 0-7 lanes<16, 8-15 >=16)
#pragma unroll
  for (int v = 0; v < 8; ++v) {
    float mv = minv[v];
    int   mi = mini[v];
#pragma unroll
    for (int mask = 1; mask < 16; mask <<= 1) {
      const float ov = __shfl_xor(mv, mask, 32);
      const int   oi = __shfl_xor(mi, mask, 32);
      if (ov < mv || (ov == mv && oi < mi)) { mv = ov; mi = oi; }
    }
    if (col == 0) {
      const int tl = w * 16 + rowoff + v;
      ldsIdx[tl] = mi;
      idxOut[t0 + tl] = (float)mi;
    }
  }
  __syncthreads();

  // ---- Phase 3: gather codebook row (f32), write out, squared error
  float lsum = 0.f;
  {
    const int k = ldsIdx[i];
    const float* crow = cb + (size_t)k * Dn;
    for (int it = 0; it < Dn / 2; ++it) {
      const int d = it * 2 + dg;
      const size_t off = xbase + (size_t)d * HWn + i;
      const float q = crow[d];
      const float diff = x[off] - q;   // x re-read is L2-resident (33.5MB < 192MB)
      lsum += diff * diff;
      out[off] = q;                    // straight-through forward value == quant
    }
  }
  ldsPS[tid] = lsum;
  __syncthreads();
  for (int s = 128; s > 0; s >>= 1) {
    if (tid < s) ldsPS[tid] += ldsPS[tid + s];
    __syncthreads();
  }
  if (tid == 0) lossPart[blockIdx.x] = ldsPS[0];
}

// ---------------------------------------------------------------------------
// Finalize: deterministic reduce of 256 block partials -> both loss scalars.
// ---------------------------------------------------------------------------
__global__ void __launch_bounds__(256) vq_finalize(const float* __restrict__ lossPart,
                                                   float* __restrict__ lossOut) {
  __shared__ float red[256];
  const int tid = threadIdx.x;
  red[tid] = lossPart[tid];
  __syncthreads();
  for (int s = 128; s > 0; s >>= 1) {
    if (tid < s) red[tid] += red[tid + s];
    __syncthreads();
  }
  if (tid == 0) {
    const float m = red[0] / (float)OUT_ELEMS;
    lossOut[0] = m;  // codebook_loss
    lossOut[1] = m;  // commitment_loss (numerically identical forward value)
  }
}

extern "C" void kernel_launch(void* const* d_in, const int* in_sizes, int n_in,
                              void* d_out, int out_size, void* d_ws, size_t ws_size,
                              hipStream_t stream) {
  (void)in_sizes; (void)n_in; (void)out_size; (void)ws_size;
  const float* x  = (const float*)d_in[0];   // (B, D, H, W) f32
  const float* cb = (const float*)d_in[1];   // (K, D) f32

  char* ws = (char*)d_ws;
  unsigned short* cbB = (unsigned short*)ws;                         // 512 KB frags
  float* csq      = (float*)(ws + (size_t)Kn * Dn * 2);              // 4 KB
  float* lossPart = (float*)(ws + (size_t)Kn * Dn * 2 + Kn * 4);     // 1 KB

  float* out     = (float*)d_out;            // [OUT_ELEMS]
  float* idxOut  = out + OUT_ELEMS;          // [Nn] (indices as f32)
  float* lossOut = idxOut + Nn;              // [2]

  vq_prep<<<KT, 256, 0, stream>>>(cb, cbB, csq);
  vq_main<<<Nn / TOK_PER_BLOCK, 256, 0, stream>>>(x, cb, cbB, csq, out, idxOut, lossPart);
  vq_finalize<<<1, 256, 0, stream>>>(lossPart, lossOut);
}
